// IPDF_Encoder_75943611728458
// MI455X (gfx1250) — matrix operand order repaired and checked by reading, not compiled, not run on hardware
//
#include <hip/hip_runtime.h>

typedef __bf16 bf16_t;
typedef __attribute__((ext_vector_type(16))) __bf16 v16bf;
typedef __attribute__((ext_vector_type(8)))  float  v8f;
typedef __attribute__((ext_vector_type(4)))  unsigned int u32x4;
typedef __attribute__((ext_vector_type(8)))  int i32x8;
typedef __attribute__((ext_vector_type(4)))  int i32x4;

#define V_LEN 32768
#define HID   256
#define NT    128
#define NTILES (V_LEN / NT)   // 256 column tiles per sample

// K position inside a 32-wide K block for 16-bit A fragments (ISA 7.12.2):
// lanes 0-15 hold K 0..7 (vgpr0..3) and 16..23 (vgpr4..7); lanes 16-31 are +8.
__device__ inline int a_kpos(int v, int h, int pos) {
    return (v < 4 ? 2 * v : 16 + 2 * (v - 4)) + 8 * h + pos;
}

// ---------------------------------------------------------------------------
// Prep: convert all experts' W2 (f32, [8][256][256]) into bf16 WMMA A-fragment
// order in scratch: frag[((e*8+kb)*16+m)*512 + lane*16 + j]
// ---------------------------------------------------------------------------
__global__ __launch_bounds__(256) void prep_w2_frags(const float* __restrict__ w2,
                                                     bf16_t* __restrict__ frag) {
    int idx  = blockIdx.x * 256 + threadIdx.x;      // 524288 total elements
    int j    = idx & 15;
    int lane = (idx >> 4) & 31;
    int m    = (idx >> 9) & 15;
    int kb   = (idx >> 13) & 7;
    int e    = idx >> 16;
    int h    = lane >> 4;
    int row  = m * 16 + (lane & 15);
    int k    = kb * 32 + 16 * h + j;
    frag[idx] = (bf16_t)w2[(e * HID + row) * HID + k];
}

// ---------------------------------------------------------------------------
// TDM: DMA the 6x128 f32 x-tile (row stride V_LEN) into LDS.
// D# bit layout per CDNA5 ISA §8.3 (group0) / §8.4 (group1).
// ---------------------------------------------------------------------------
__device__ inline void tdm_load_x_tile(const float* gsrc, unsigned lds_addr) {
    unsigned long long ga = (unsigned long long)(size_t)gsrc;
    u32x4 g0;
    g0[0] = 1u;                                     // count=1, user descriptor
    g0[1] = lds_addr;                               // lds_addr [63:32]
    g0[2] = (unsigned)(ga & 0xFFFFFFFFull);         // global_addr [95:64]
    g0[3] = (unsigned)((ga >> 32) & 0x01FFFFFFull)  // global_addr [120:96]
          | (2u << 30);                             // type=2 ("image")
    i32x8 g1;
    g1[0] = (int)(2u << 16);      // workgroup_mask=0, data_size=2 (4B)
    g1[1] = (int)0x80000000u;     // tensor_dim0 = 32768 (low 16 in bits [31:16])
    g1[2] = (int)(6u << 16);      // tensor_dim0 hi=0 | tensor_dim1 = 6
    g1[3] = (int)(128u << 16);    // tensor_dim1 hi=0 | tile_dim0 = 128
    g1[4] = 6;                    // tile_dim1 = 6, tile_dim2 = 0 (2D tile)
    g1[5] = (int)32768;           // tensor_dim0_stride low 32 (= V_LEN elements)
    g1[6] = 0;                    // stride0 hi | stride1 lo (unused for 2D)
    g1[7] = 0;
    i32x4 z4 = (i32x4)0;          // groups 2/3 unused (2D tensor)
#if __has_include(<hip/amd_detail/amd_gfx1250_TDM.h>)
    i32x8 z8 = (i32x8)0;
    __builtin_amdgcn_tensor_load_to_lds(g0, g1, z4, z4, z8, 0);
#else
    __builtin_amdgcn_tensor_load_to_lds(g0, g1, z4, z4, 0);
#endif
}

// ---------------------------------------------------------------------------
// Fused 3-layer MLP. Grid: (NTILES, B). Block: 256 threads = 8 waves.
// ---------------------------------------------------------------------------
__global__ __launch_bounds__(256) void fused_mlp(const float* __restrict__ x,
                                                 const int*   __restrict__ obj_idx,
                                                 const float* __restrict__ w1,
                                                 const float* __restrict__ b1,
                                                 const float* __restrict__ b2,
                                                 const float* __restrict__ w3,
                                                 const float* __restrict__ b3,
                                                 const bf16_t* __restrict__ w2f,
                                                 float* __restrict__ out) {
    __shared__ float  s_x[6][NT];                       // 3 KB (TDM destination)
    __shared__ float  s_w1[HID * 6];                    // 6 KB
    __shared__ float  s_b1[HID];                        // 1 KB
    __shared__ __align__(32) bf16_t s_B[8][8][32][16];  // 64 KB: kb, ntile, lane, j
    __shared__ float  s_red[NT];                        // 0.5 KB

    const int tid  = threadIdx.x;
    const int lane = tid & 31;
    const int wave = tid >> 5;
    const int b    = blockIdx.y;
    const int col0 = blockIdx.x * NT;
    const int e    = obj_idx[b];

    // ---- wave 0: kick off TDM for the x tile; all waves: stage w1/b1 ----
    if (wave == 0) {
        tdm_load_x_tile(x + (size_t)(b * 6) * V_LEN + col0,
                        (unsigned)(size_t)&s_x[0][0]);
    }
    for (int i = tid; i < HID * 6; i += 256) s_w1[i] = w1[e * HID * 6 + i];
    s_b1[tid] = b1[e * HID + tid];
    if (tid < NT) s_red[tid] = 0.0f;
    if (wave == 0) __builtin_amdgcn_s_wait_tensorcnt(0);
    __syncthreads();

    // ---- layer 1: h1 = relu(W1 x + b1), stored as bf16 B-fragments ----
    {
        const int n   = tid >> 5;           // column tile 0..7
        const int l   = tid & 31;
        const int g   = l >> 4;             // K half within 32-block
        const int col = n * 16 + (l & 15);
        float xc[6];
        #pragma unroll
        for (int i = 0; i < 6; ++i) xc[i] = s_x[i][col];
        for (int kb = 0; kb < 8; ++kb) {
            const int k0 = kb * 32 + g * 16;
            v16bf hv;
            #pragma unroll
            for (int j = 0; j < 16; ++j) {
                const int k = k0 + j;
                float h = s_b1[k];
                #pragma unroll
                for (int i = 0; i < 6; ++i) h += s_w1[k * 6 + i] * xc[i];
                hv[j] = (bf16_t)fmaxf(h, 0.0f);
            }
            *reinterpret_cast<v16bf*>(&s_B[kb][n][l][0]) = hv;
        }
    }
    __syncthreads();

    // ---- layer 2: H2 = W2 @ H1 via v_wmma_f32_16x16x32_bf16 ----
    // wave -> 4 m-tiles x 4 n-tiles (A and B fragments each reused 4x)
    const int m0 = (wave & 3) * 4;          // first of 4 row tiles
    const int n0 = (wave >> 2) * 4;         // first of 4 col tiles
    v8f acc[4][4];
    #pragma unroll
    for (int mi = 0; mi < 4; ++mi)
        #pragma unroll
        for (int n = 0; n < 4; ++n) acc[mi][n] = (v8f)0.0f;

    const bf16_t* fragbase = w2f + (size_t)e * 8 * 16 * 512;
    for (int kb = 0; kb < 8; ++kb) {
        const bf16_t* fb = fragbase + (size_t)kb * 16 * 512;
        if (kb < 7)   // pull next K-block's A fragments toward the WGP cache
            __builtin_prefetch(fb + 16 * 512 + m0 * 512 + lane * 16, 0, 3);
        v16bf a[4];
        #pragma unroll
        for (int mi = 0; mi < 4; ++mi)
            a[mi] = *reinterpret_cast<const v16bf*>(fb + (m0 + mi) * 512 + lane * 16);
        #pragma unroll
        for (int n = 0; n < 4; ++n) {
            const v16bf bv = *reinterpret_cast<const v16bf*>(&s_B[kb][n0 + n][lane][0]);
            #pragma unroll
            for (int mi = 0; mi < 4; ++mi)
                acc[mi][n] = __builtin_amdgcn_wmma_f32_16x16x32_bf16(
                    false, a[mi], false, bv, (short)0, acc[mi][n], false, false);
        }
    }

    // ---- layer 3 fused epilogue: relu(H2 + b2) dotted with w3, reduce ----
    float pc[4];
    #pragma unroll
    for (int n = 0; n < 4; ++n) pc[n] = 0.0f;
    const int half = lane >> 4;
    #pragma unroll
    for (int mi = 0; mi < 4; ++mi) {
        #pragma unroll
        for (int r = 0; r < 8; ++r) {
            const int   row  = (m0 + mi) * 16 + r + 8 * half; // C/D layout (ISA 7.12.2)
            const float bias = b2[e * HID + row];
            const float wrow = w3[e * HID + row];
            #pragma unroll
            for (int n = 0; n < 4; ++n)
                pc[n] += wrow * fmaxf(acc[mi][n][r] + bias, 0.0f);
        }
    }
    #pragma unroll
    for (int n = 0; n < 4; ++n)
        atomicAdd(&s_red[(n0 + n) * 16 + (lane & 15)], pc[n]);  // ds_add_f32
    __syncthreads();

    if (tid < NT)
        out[(size_t)b * V_LEN + col0 + tid] = s_red[tid] + b3[e];
}

// ---------------------------------------------------------------------------
extern "C" void kernel_launch(void* const* d_in, const int* in_sizes, int n_in,
                              void* d_out, int out_size, void* d_ws, size_t ws_size,
                              hipStream_t stream) {
    const float* x  = (const float*)d_in[0];
    const int*   oi = (const int*)d_in[1];
    const float* w1 = (const float*)d_in[2];
    const float* b1 = (const float*)d_in[3];
    const float* w2 = (const float*)d_in[4];
    const float* b2 = (const float*)d_in[5];
    const float* w3 = (const float*)d_in[6];
    const float* b3 = (const float*)d_in[7];
    float*  out   = (float*)d_out;
    bf16_t* frags = (bf16_t*)d_ws;   // 1 MB of bf16 W2 fragments

    prep_w2_frags<<<2048, 256, 0, stream>>>(w2, frags);
    dim3 grid(NTILES, 16);
    fused_mlp<<<grid, 256, 0, stream>>>(x, oi, w1, b1, b2, w3, b3, frags, out);
    (void)in_sizes; (void)n_in; (void)out_size; (void)ws_size;
}
